// TransformerEncoderLayer_34170759807690
// MI455X (gfx1250) — matrix operand order repaired and checked by reading, not compiled, not run on hardware
//
#include <hip/hip_runtime.h>
#include <hip/hip_bf16.h>

typedef __bf16 bf16;
typedef __attribute__((ext_vector_type(4)))  __bf16 v4bf;
typedef __attribute__((ext_vector_type(16))) __bf16 v16bf;
typedef __attribute__((ext_vector_type(8)))  float  v8f;

// ---------------------------------------------------------------------------
// WMMA fragment helpers (CDNA5 wave32 layouts, per CDNA5 ISA)
// A 16x32 bf16: lane L holds row M=L%16; elems 0..7 -> K in [lo,lo+8),
//               elems 8..15 -> K in [lo+16,lo+24), lo = (L<16 ? 0 : 8).
// B 32x16 bf16 (stored transposed as [N][K]): lane holds col N=L%16,
//               elems 0..7 -> K in [lo,lo+8), elems 8..15 -> K in [lo+16,lo+24), lo = (L<16 ? 0 : 8).
// C 16x16 f32: lane holds col N=L%16; VGPR v holds row M = v + 8*(L>=16).
// ---------------------------------------------------------------------------
__device__ __forceinline__ v16bf frag_a(const bf16* base, int ld) {
  const int lane = threadIdx.x & 31;
  const bf16* r = base + (lane & 15) * ld + ((lane < 16) ? 0 : 8);
  union { v16bf v; uint4 q[2]; } u;
  u.q[0] = *(const uint4*)(r);
  u.q[1] = *(const uint4*)(r + 16);
  return u.v;
}

__device__ __forceinline__ v16bf frag_b(const bf16* base, int ld) {
  const int lane = threadIdx.x & 31;
  const bf16* r = base + (lane & 15) * ld + ((lane < 16) ? 0 : 8);
  union { v16bf v; uint4 q[2]; } u;
  u.q[0] = *(const uint4*)(r);
  u.q[1] = *(const uint4*)(r + 16);
  return u.v;
}

__device__ __forceinline__ v8f wmma_bf16(v16bf a, v16bf b, v8f c) {
  return __builtin_amdgcn_wmma_f32_16x16x32_bf16(
      /*neg_a=*/false, a, /*neg_b=*/false, b,
      /*c_mod=*/(short)0, c, /*reuse_a=*/false, /*reuse_b=*/false);
}

// ---------------------------------------------------------------------------
// Projection GEMM: dst = (A[Mx1024] @ W[1024x1024] + bias) * scale  -> bf16
// mode 0: QKV -> [B,H,S,D] head layout.  mode 1: pos -> [H, 1024, D].
// Block = 128 threads (4 waves), tile 64x64, K-chunk 32, software-pipelined:
// vectorized float4 global loads of chunk i+1 issue while WMMAs consume
// chunk i from LDS.
// ---------------------------------------------------------------------------
__global__ __launch_bounds__(128) void proj_gemm(
    const float* __restrict__ A, const float* __restrict__ W,
    const float* __restrict__ bias, bf16* __restrict__ dst,
    int mode, float scale) {
  __shared__ __attribute__((aligned(16))) bf16 As[64 * 40];
  __shared__ __attribute__((aligned(16))) bf16 Bs[64 * 40];  // W^T tile: [n][k]

  const int n0 = blockIdx.x * 64;
  const int m0 = blockIdx.y * 64;
  const int t = threadIdx.x;
  const int lane = t & 31, w = t >> 5, half = (lane >> 4);

  float4 ar[4], wr[4];

  auto stage_load = [&](int k0) {
    #pragma unroll
    for (int r = 0; r < 4; ++r) {
      int f = t + r * 128;                  // 0..511 (x4 elements each)
      int ia = f >> 3, ja = (f & 7) * 4;    // A: 64 rows x 32 k
      ar[r] = *(const float4*)&A[(size_t)(m0 + ia) * 1024 + (k0 + ja)];
      int kw = f >> 4, jw = (f & 15) * 4;   // W: 32 k x 64 n, coalesced in n
      wr[r] = *(const float4*)&W[(size_t)(k0 + kw) * 1024 + (n0 + jw)];
    }
  };
  auto stage_store = [&]() {
    #pragma unroll
    for (int r = 0; r < 4; ++r) {
      int f = t + r * 128;
      int ia = f >> 3, ja = (f & 7) * 4;
      v4bf av = {(bf16)ar[r].x, (bf16)ar[r].y, (bf16)ar[r].z, (bf16)ar[r].w};
      *(v4bf*)&As[ia * 40 + ja] = av;       // 8B aligned packed store
      int kw = f >> 4, jw = (f & 15) * 4;   // transpose W into [n][k]
      Bs[(jw + 0) * 40 + kw] = (bf16)wr[r].x;
      Bs[(jw + 1) * 40 + kw] = (bf16)wr[r].y;
      Bs[(jw + 2) * 40 + kw] = (bf16)wr[r].z;
      Bs[(jw + 3) * 40 + kw] = (bf16)wr[r].w;
    }
  };

  v8f acc[4] = {{}, {}, {}, {}};
  stage_load(0);
  for (int k0 = 0; k0 < 1024; k0 += 32) {
    stage_store();
    __syncthreads();
    if (k0 + 32 < 1024) stage_load(k0 + 32);   // overlap with WMMAs below

    v16bf a = frag_a(&As[(w * 16) * 40], 40);
    #pragma unroll
    for (int c = 0; c < 4; ++c) {
      v16bf b = frag_b(&Bs[(c * 16) * 40], 40);
      acc[c] = wmma_bf16(a, b, acc[c]);
    }
    __syncthreads();
  }

  // epilogue: bias, scale, layout remap, store bf16
  #pragma unroll
  for (int c = 0; c < 4; ++c) {
    int n = n0 + c * 16 + (lane & 15);
    float bvv = bias ? bias[n] : 0.0f;
    #pragma unroll
    for (int v = 0; v < 8; ++v) {
      int m = m0 + w * 16 + v + 8 * half;
      float val = (acc[c][v] + bvv) * scale;
      size_t idx;
      if (mode == 0) {
        int bb = m >> 11, s = m & 2047, hh = n >> 6, d = n & 63;
        idx = ((size_t)(bb * 16 + hh) * 2048 + s) * 64 + d;
      } else {
        idx = ((size_t)(n >> 6) * 1024 + m) * 64 + (n & 63);
      }
      dst[idx] = (bf16)val;
    }
  }
}

// ---------------------------------------------------------------------------
// Fused flash-attention with DeBERTa relative-position band trick.
// Grid (32,16,2) = (q-tiles, H, B); 128 threads = 4 waves; each wave owns a
// 16(q) x 64 strip.  Per k-tile: QK^T, banded c2p / p2c via WMMA + LDS gather
// at j = (q-Q0)-(k-K0)+63, online softmax, P@V.  All staging is b128
// vectorized; next tile prefetched into cache during compute.
// ---------------------------------------------------------------------------
__global__ __launch_bounds__(128) void deberta_attn(
    const bf16* __restrict__ Qh, const bf16* __restrict__ Kh,
    const bf16* __restrict__ Vh, const bf16* __restrict__ PKh,
    const bf16* __restrict__ PQh, float* __restrict__ out) {
  __shared__ __attribute__((aligned(16))) bf16 Qt[64 * 72];
  __shared__ __attribute__((aligned(16))) bf16 Kt[64 * 72];
  __shared__ __attribute__((aligned(16))) bf16 Vt[64 * 72];   // V^T : [d][k]
  __shared__ __attribute__((aligned(16))) bf16 Pb[128 * 72];  // pos band
  __shared__ __attribute__((aligned(16))) bf16 Cb[64 * 136];  // band prods / P

  const int qt = blockIdx.x, h = blockIdx.y, b = blockIdx.z;
  const int Q0 = qt * 64;
  const int t = threadIdx.x, lane = t & 31, w = t >> 5, half = lane >> 4;
  const size_t headoff = (size_t)(b * 16 + h) * 2048 * 64;
  const size_t posoff  = (size_t)h * 1024 * 64;

  // load Q tile (persistent), vectorized 8-bf16 chunks
  #pragma unroll
  for (int r = 0; r < 4; ++r) {
    int f = t + r * 128;            // 0..511
    int i = f >> 3, d = (f & 7) * 8;
    *(uint4*)&Qt[i * 72 + d] =
        *(const uint4*)&Qh[headoff + (size_t)(Q0 + i) * 64 + d];
  }

  float mrow[8], lrow[8];
  v8f O[4] = {{}, {}, {}, {}};
  #pragma unroll
  for (int v = 0; v < 8; ++v) { mrow[v] = -1e30f; lrow[v] = 0.0f; }

  for (int kt = 0; kt < 32; ++kt) {
    const int K0 = kt * 64;
    const int lo = Q0 - K0 + 449;   // band base: delta = q-k+512, j = delta-lo

    // stage K tile + V^T tile (vector loads; transpose unpack for V)
    #pragma unroll
    for (int r = 0; r < 4; ++r) {
      int f = t + r * 128;
      int i = f >> 3, d = (f & 7) * 8;
      *(uint4*)&Kt[i * 72 + d] =
          *(const uint4*)&Kh[headoff + (size_t)(K0 + i) * 64 + d];
      union { uint4 q; bf16 e[8]; } uv;
      uv.q = *(const uint4*)&Vh[headoff + (size_t)(K0 + i) * 64 + d];
      #pragma unroll
      for (int e2 = 0; e2 < 8; ++e2) Vt[(d + e2) * 72 + i] = uv.e[e2];
    }
    // stage pos_k band (row clamping implements the clip() for free)
    #pragma unroll
    for (int r = 0; r < 8; ++r) {
      int f = t + r * 128;          // 0..1023
      int j = f >> 3, d = (f & 7) * 8;
      int rr = lo + j; rr = rr < 0 ? 0 : (rr > 1023 ? 1023 : rr);
      *(uint4*)&Pb[j * 72 + d] =
          *(const uint4*)&PKh[posoff + (size_t)rr * 64 + d];
    }
    // prefetch next K/V tiles into cache (global_prefetch_b8)
    if (kt + 1 < 32) {
      const bf16* nk = Kh + headoff + (size_t)(K0 + 64) * 64;
      const bf16* nv = Vh + headoff + (size_t)(K0 + 64) * 64;
      __builtin_prefetch(nk + t * 32, 0, 1);
      __builtin_prefetch(nv + t * 32, 0, 1);
    }
    __syncthreads();

    v16bf aq0 = frag_a(&Qt[(w * 16) * 72], 72);
    v16bf aq1 = frag_a(&Qt[(w * 16) * 72 + 32], 72);

    // c2p band:  Cb[qloc][j] = Q[q] . pos_k[clip(lo+j)]
    #pragma unroll
    for (int c = 0; c < 8; ++c) {
      v16bf b0 = frag_b(&Pb[(c * 16) * 72], 72);
      v16bf b1 = frag_b(&Pb[(c * 16) * 72 + 32], 72);
      v8f cc = {};
      cc = wmma_bf16(aq0, b0, cc);
      cc = wmma_bf16(aq1, b1, cc);
      #pragma unroll
      for (int v = 0; v < 8; ++v)
        Cb[(w * 16 + v + 8 * half) * 136 + c * 16 + (lane & 15)] = (bf16)cc[v];
    }
    __syncthreads();

    // QK^T
    v8f s[4];
    #pragma unroll
    for (int c = 0; c < 4; ++c) {
      v16bf b0 = frag_b(&Kt[(c * 16) * 72], 72);
      v16bf b1 = frag_b(&Kt[(c * 16) * 72 + 32], 72);
      v8f sc = {};
      sc = wmma_bf16(aq0, b0, sc);
      sc = wmma_bf16(aq1, b1, sc);
      s[c] = sc;
    }
    // c2p gather: j = qloc - kloc + 63 in [0,126]
    #pragma unroll
    for (int c = 0; c < 4; ++c) {
      int kloc = c * 16 + (lane & 15);
      #pragma unroll
      for (int v = 0; v < 8; ++v) {
        int qloc = w * 16 + v + 8 * half;
        s[c][v] += (float)Cb[qloc * 136 + (qloc - kloc + 63)];
      }
    }
    // stage pos_q band (Pb fully consumed by c2p WMMAs before the barrier)
    #pragma unroll
    for (int r = 0; r < 8; ++r) {
      int f = t + r * 128;
      int j = f >> 3, d = (f & 7) * 8;
      int rr = lo + j; rr = rr < 0 ? 0 : (rr > 1023 ? 1023 : rr);
      *(uint4*)&Pb[j * 72 + d] =
          *(const uint4*)&PQh[posoff + (size_t)rr * 64 + d];
    }
    __syncthreads();

    // p2c band:  Cb[kloc][j] = K[k] . pos_q[clip(lo+j)]
    v16bf ak0 = frag_a(&Kt[(w * 16) * 72], 72);
    v16bf ak1 = frag_a(&Kt[(w * 16) * 72 + 32], 72);
    #pragma unroll
    for (int c = 0; c < 8; ++c) {
      v16bf b0 = frag_b(&Pb[(c * 16) * 72], 72);
      v16bf b1 = frag_b(&Pb[(c * 16) * 72 + 32], 72);
      v8f cc = {};
      cc = wmma_bf16(ak0, b0, cc);
      cc = wmma_bf16(ak1, b1, cc);
      #pragma unroll
      for (int v = 0; v < 8; ++v)
        Cb[(w * 16 + v + 8 * half) * 136 + c * 16 + (lane & 15)] = (bf16)cc[v];
    }
    __syncthreads();
    // p2c gather + final 1/sqrt(D) scaling
    #pragma unroll
    for (int c = 0; c < 4; ++c) {
      int kloc = c * 16 + (lane & 15);
      #pragma unroll
      for (int v = 0; v < 8; ++v) {
        int qloc = w * 16 + v + 8 * half;
        s[c][v] = (s[c][v] + (float)Cb[kloc * 136 + (qloc - kloc + 63)]) * 0.125f;
      }
    }
    __syncthreads();  // everyone done reading Cb before P overwrite

    // ---- online softmax (rows live in 16-lane halves; shfl_xor <16 stays in-half)
    float mnew[8], alpha[8];
    #pragma unroll
    for (int v = 0; v < 8; ++v) {
      float mv = fmaxf(fmaxf(s[0][v], s[1][v]), fmaxf(s[2][v], s[3][v]));
      #pragma unroll
      for (int off = 1; off < 16; off <<= 1)
        mv = fmaxf(mv, __shfl_xor(mv, off, 32));
      mnew[v] = fmaxf(mrow[v], mv);
      alpha[v] = __expf(mrow[v] - mnew[v]);
      mrow[v] = mnew[v];
    }
    float lsum[8];
    #pragma unroll
    for (int v = 0; v < 8; ++v) lsum[v] = 0.0f;
    #pragma unroll
    for (int c = 0; c < 4; ++c) {
      #pragma unroll
      for (int v = 0; v < 8; ++v) {
        float p = __expf(s[c][v] - mnew[v]);
        lsum[v] += p;
        Cb[(w * 16 + v + 8 * half) * 136 + c * 16 + (lane & 15)] = (bf16)p;
      }
    }
    #pragma unroll
    for (int v = 0; v < 8; ++v) {
      float lv = lsum[v];
      #pragma unroll
      for (int off = 1; off < 16; off <<= 1) lv += __shfl_xor(lv, off, 32);
      lrow[v] = lrow[v] * alpha[v] + lv;
      O[0][v] *= alpha[v]; O[1][v] *= alpha[v];
      O[2][v] *= alpha[v]; O[3][v] *= alpha[v];
    }
    __syncthreads();

    // O += P @ V   (P rows = this wave's q rows, K-dim = 64 keys)
    v16bf ap0 = frag_a(&Cb[(w * 16) * 136], 136);
    v16bf ap1 = frag_a(&Cb[(w * 16) * 136 + 32], 136);
    #pragma unroll
    for (int c = 0; c < 4; ++c) {
      v16bf b0 = frag_b(&Vt[(c * 16) * 72], 72);
      v16bf b1 = frag_b(&Vt[(c * 16) * 72 + 32], 72);
      O[c] = wmma_bf16(ap0, b0, O[c]);
      O[c] = wmma_bf16(ap1, b1, O[c]);
    }
    __syncthreads();
  }

  // finalize: O /= l, write [B,S,H*D] fp32
  #pragma unroll
  for (int v = 0; v < 8; ++v) {
    int q = Q0 + w * 16 + v + 8 * half;
    float inv = 1.0f / lrow[v];
    #pragma unroll
    for (int c = 0; c < 4; ++c) {
      int d = c * 16 + (lane & 15);
      out[((size_t)b * 2048 + q) * 1024 + h * 64 + d] = O[c][v] * inv;
    }
  }
}

// ---------------------------------------------------------------------------
extern "C" void kernel_launch(void* const* d_in, const int* in_sizes, int n_in,
                              void* d_out, int out_size, void* d_ws, size_t ws_size,
                              hipStream_t stream) {
  (void)in_sizes; (void)n_in; (void)out_size; (void)ws_size;
  const float* X   = (const float*)d_in[0];
  const float* RE  = (const float*)d_in[1];
  const float* Wq  = (const float*)d_in[2];
  const float* bq  = (const float*)d_in[3];
  const float* Wk  = (const float*)d_in[4];
  const float* bk  = (const float*)d_in[5];
  const float* Wv  = (const float*)d_in[6];
  const float* bv  = (const float*)d_in[7];
  const float* Wpk = (const float*)d_in[8];
  const float* Wpq = (const float*)d_in[9];
  const float* bpq = (const float*)d_in[10];
  float* out = (float*)d_out;

  // bf16 workspace: Q,K,V head-layout (8MB each) + pos_k/pos_q (2MB each) = 28MB
  bf16* Qh  = (bf16*)d_ws;
  bf16* Kh  = Qh  + (size_t)4096 * 1024;
  bf16* Vh  = Kh  + (size_t)4096 * 1024;
  bf16* PKh = Vh  + (size_t)4096 * 1024;
  bf16* PQh = PKh + (size_t)16 * 1024 * 64;

  dim3 blk(128);
  dim3 gQKV(16, 64);                     // N/64 x M/64, M = B*S = 4096
  proj_gemm<<<gQKV, blk, 0, stream>>>(X, Wq, bq, Qh, 0, 1.0f);
  proj_gemm<<<gQKV, blk, 0, stream>>>(X, Wk, bk, Kh, 0, 1.0f);
  proj_gemm<<<gQKV, blk, 0, stream>>>(X, Wv, bv, Vh, 0, 1.0f);

  dim3 gP(16, 16);                       // M = 2*SPAN = 1024
  proj_gemm<<<gP, blk, 0, stream>>>(RE, Wpk, nullptr, PKh, 1, 1.0f);
  const float pscale = 0.07216878364870323f;  // 1/sqrt(D * 3)
  proj_gemm<<<gP, blk, 0, stream>>>(RE, Wpq, bpq, PQh, 1, pscale);

  dim3 ga(32, 16, 2);                    // q-tiles x H x B
  deberta_attn<<<ga, blk, 0, stream>>>(Qh, Kh, Vh, PKh, PQh, out);
}
